// Cell2CellwithAuto_4750233829571
// MI455X (gfx1250) — compile-verified, matrix operand order checked by reading, not run on hardware
//
#include <hip/hip_runtime.h>
#include <hip/hip_bf16.h>
#include <math.h>

// ---------------------------------------------------------------------------
// Problem constants
// ---------------------------------------------------------------------------
constexpr int NPTS   = 3000;
constexpr int NP     = 3072;   // pad of NPTS to 128 multiple (24 * 128)
constexpr int IN_DIM = 2000;
constexpr int KXP    = 2048;   // pad of IN_DIM (K and decoder-N pad)
constexpr int HID    = 256;
constexpr int HEADS  = 4;
constexpr int NCLUST = 12;
constexpr int K_KNN  = 30;     // N >= 300 -> k = 30
constexpr int KA     = 10;     // int(30/3)

typedef __attribute__((ext_vector_type(16))) _Float16 v16h;
typedef __attribute__((ext_vector_type(8)))  _Float16 v8h;
typedef __attribute__((ext_vector_type(8)))  float    v8f;

// ---------------------------------------------------------------------------
// GEMM configuration: 128x128 block tile, 256 threads (8 wave32),
// wave tile 64x32 (4x2 16x16 fragments), BK=64 (two WMMA k-substeps).
// Double-buffered LDS, filled by async global->LDS DMA (ASYNCcnt-tracked),
// so tile k+1 streams in while WMMAs consume tile k.
// LDS tiles stored [row][k], row stride SK=72 halves (144B: 16B aligned,
// bank-spread for the b128 fragment loads).
// ---------------------------------------------------------------------------
constexpr int BM = 128, BN = 128, BK = 64, SK = 72, GT = 256;

constexpr unsigned EP_C16      = 1u;   // store C as f16
constexpr unsigned EP_C32      = 2u;   // store C as f32
constexpr unsigned EP_CT16     = 4u;   // store C^T as f16 (vectorized v8h)
constexpr unsigned EP_BIAS     = 8u;   // += bias[col]
constexpr unsigned EP_SOFTPLUS = 16u;  // softplus activation
constexpr unsigned EP_ACC32    = 32u;  // C32 += result

// Async copy of 16B global -> LDS (no VGPR data path, tracked by ASYNCcnt).
// LDS operand: wave-relative byte offset = low 32 bits of the generic pointer
// (ISA: flat->LDS uses addr[31:0]).
__device__ inline void async_g2l_b128(const void* gptr, void* lptr) {
  const unsigned loff = (unsigned)(uintptr_t)lptr;
  asm volatile("global_load_async_to_lds_b128 %0, %1, off"
               :: "v"(loff), "v"(gptr) : "memory");
}
__device__ inline void wait_asynccnt0() {
  asm volatile("s_wait_asynccnt 0x0" ::: "memory");
}

// Fragment load from LDS tile [row][k], stride SK.  CDNA5 16-bit A/B layout:
// lane L -> row (L&15), elements e -> K = (e>>3)*16 + 8*(L>>4) + (e&7).
__device__ inline v16h frag_ld(const _Float16* __restrict__ base, int row0, int kOff) {
  const int lane = threadIdx.x & 31;
  const int r    = lane & 15;
  const int kb   = ((lane >> 4) & 1) * 8;
  const _Float16* p = base + (row0 + r) * SK + kOff + kb;
  v8h lo = *(const v8h*)(p);
  v8h hi = *(const v8h*)(p + 16);
  v16h o;
#pragma unroll
  for (int e = 0; e < 8; ++e) { o[e] = lo[e]; o[e + 8] = hi[e]; }
  return o;
}

// Full-tile GEMM:  C(MxN) = A(MxK) @ Bt(NxK)^T.  All dims multiples of the
// tile; all pointers 16B aligned; zero bounds checks anywhere.
template <unsigned EPI>
__global__ __launch_bounds__(GT)
void wmma_gemm(const _Float16* __restrict__ A, const _Float16* __restrict__ Bt,
               const float* __restrict__ bias,
               _Float16* __restrict__ C16, float* __restrict__ C32,
               _Float16* __restrict__ Ct16,
               int K, int lda, int ldbt, int ldc, int ldct)
{
  __shared__ __align__(16) _Float16 sA[2][BM * SK];
  __shared__ __align__(16) _Float16 sB[2][BN * SK];

  const int bm   = blockIdx.y * BM;
  const int bn   = blockIdx.x * BN;
  const int tid  = threadIdx.x;
  const int wave = tid >> 5;
  const int wm   = (wave & 1) * 64;   // 2 waves across M
  const int wn   = (wave >> 1) * 32;  // 4 waves across N

  v8f acc[4][2] = {};

  // Per-thread staging chunks: 128x64 halves = 1024 v8h chunks per tile.
  auto stage_async = [&](int k0, int buf) {
#pragma unroll
    for (int it = 0; it < 4; ++it) {
      const int c   = tid + GT * it;
      const int row = c >> 3;
      const int k8  = (c & 7) * 8;
      async_g2l_b128(A  + (size_t)(bm + row) * lda  + k0 + k8,
                     &sA[buf][row * SK + k8]);
      async_g2l_b128(Bt + (size_t)(bn + row) * ldbt + k0 + k8,
                     &sB[buf][row * SK + k8]);
    }
  };

  // Prologue: DMA first tile, wait, sync.
  stage_async(0, 0);
  wait_asynccnt0();
  __syncthreads();

  int buf = 0;
  for (int k0 = 0; k0 < K; k0 += BK) {
    // Prefetch next tile into the other buffer (it was last read before the
    // previous barrier, so overwriting it now is race-free).
    if (k0 + BK < K) stage_async(k0 + BK, buf ^ 1);

#pragma unroll
    for (int ks = 0; ks < 2; ++ks) {
      v16h af[4], bf[2];
#pragma unroll
      for (int i = 0; i < 4; ++i) af[i] = frag_ld(sA[buf], wm + i * 16, ks * 32);
#pragma unroll
      for (int j = 0; j < 2; ++j) bf[j] = frag_ld(sB[buf], wn + j * 16, ks * 32);
#pragma unroll
      for (int i = 0; i < 4; ++i)
#pragma unroll
        for (int j = 0; j < 2; ++j)
          acc[i][j] = __builtin_amdgcn_wmma_f32_16x16x32_f16(
              false, af[i], false, bf[j], (short)0, acc[i][j], false, false);
    }

    wait_asynccnt0();      // prefetch landed in LDS
    __syncthreads();       // all waves done reading current buffer
    buf ^= 1;
  }

  // Epilogue. D layout: lane L -> col = (L&15), VGPR v -> row = v + 8*(L>>4).
  const int lane = tid & 31;
  const int cn   = lane & 15;
  const int mB   = (lane >> 4) * 8;
#pragma unroll
  for (int i = 0; i < 4; ++i) {
    const int row0 = bm + wm + i * 16 + mB;
#pragma unroll
    for (int j = 0; j < 2; ++j) {
      const int col = bn + wn + j * 16 + cn;
      float vals[8];
#pragma unroll
      for (int v = 0; v < 8; ++v) {
        float x = acc[i][j][v];
        if (EPI & EP_BIAS)     x += bias[col];
        if (EPI & EP_SOFTPLUS) x = (x > 20.f) ? x : log1pf(__expf(x));
        vals[v] = x;
      }
      if (EPI & EP_C16) {
#pragma unroll
        for (int v = 0; v < 8; ++v)
          C16[(size_t)(row0 + v) * ldc + col] = (_Float16)vals[v];
      }
      if (EPI & EP_C32) {
#pragma unroll
        for (int v = 0; v < 8; ++v)
          C32[(size_t)(row0 + v) * ldc + col] = vals[v];
      }
      if (EPI & EP_ACC32) {
#pragma unroll
        for (int v = 0; v < 8; ++v)
          C32[(size_t)(row0 + v) * ldc + col] += vals[v];
      }
      if (EPI & EP_CT16) {                       // vectorized transposed store
        v8h t;
#pragma unroll
        for (int v = 0; v < 8; ++v) t[v] = (_Float16)vals[v];
        *(v8h*)(Ct16 + (size_t)col * ldct + row0) = t;
      }
    }
  }
}

// ---------------------------------------------------------------------------
// Prep / elementwise kernels
// ---------------------------------------------------------------------------
__global__ void pad16_kernel(const float* __restrict__ in, _Float16* __restrict__ out,
                             int R, int C, int ldin, int Cp) {
  const int c = blockIdx.x * 256 + threadIdx.x;
  const int r = blockIdx.y;
  if (c < Cp)
    out[(size_t)r * Cp + c] = (r < R && c < C) ? (_Float16)in[(size_t)r * ldin + c]
                                               : (_Float16)0.f;
}

// out[c][r] (Cp x Rp, f16) = in[r][c] zero-padded (transposed convert)
__global__ void padT16_kernel(const float* __restrict__ in, _Float16* __restrict__ out,
                              int R, int C, int ldin, int Rp) {
  const int r = blockIdx.x * 256 + threadIdx.x;  // contiguous out column
  const int c = blockIdx.y;                      // out row
  if (r < Rp)
    out[(size_t)c * Rp + r] = (r < R && c < C) ? (_Float16)in[(size_t)r * ldin + c]
                                               : (_Float16)0.f;
}

__global__ void padbias_kernel(const float* __restrict__ in, float* __restrict__ out,
                               int C, int Cp) {
  const int c = blockIdx.x * 256 + threadIdx.x;
  if (c < Cp) out[c] = (c < C) ? in[c] : 0.f;
}

__global__ void zero16_kernel(_Float16* __restrict__ p, size_t total) {
  size_t i = (size_t)blockIdx.x * 256 + threadIdx.x;
  const size_t stride = (size_t)gridDim.x * 256;
  for (; i < total; i += stride) p[i] = (_Float16)0.f;
}

__global__ void zero32_kernel(float* __restrict__ p, size_t total) {
  size_t i = (size_t)blockIdx.x * 256 + threadIdx.x;
  const size_t stride = (size_t)gridDim.x * 256;
  for (; i < total; i += stride) p[i] = 0.f;
}

__global__ void copy_out_kernel(const float* __restrict__ in, float* __restrict__ out,
                                int C, int ldin) {
  const int c = blockIdx.x * 256 + threadIdx.x;
  const int r = blockIdx.y;
  if (c < C) out[(size_t)r * C + c] = in[(size_t)r * ldin + c];
}

// sq[i] = ||q_i||^2
__global__ __launch_bounds__(256)
void sqnorm_kernel(const _Float16* __restrict__ q, float* __restrict__ sq, int k) {
  __shared__ float red[256];
  const int i = blockIdx.x, t = threadIdx.x;
  float v = 0.f;
  if (t < k) { const float x = (float)q[(size_t)i * k + t]; v = x * x; }
  red[t] = v; __syncthreads();
  for (int s = 128; s > 0; s >>= 1) { if (t < s) red[t] += red[t + s]; __syncthreads(); }
  if (t == 0) sq[i] = red[0];
}

// Per-row Markov affinity (in-place G -> aff).  Exact 11th/30th smallest
// distance by iterative min-extraction in LDS; kNN threshold via monotone
// equivalence (30th-largest aff == aff of 30th-smallest dist); diag = 1.
__global__ __launch_bounds__(256)
void markov_rows(float* __restrict__ G, const float* __restrict__ sq, int n, int ld) {
  __shared__ float row[NP];
  __shared__ float redv[256];
  __shared__ int   redc[256];
  const int i = blockIdx.x, tid = threadIdx.x;
  const float sqi = sq[i];

  for (int j = tid; j < NP; j += 256) {
    float d = 3.4e38f;
    if (j < n) {
      const float d2 = fmaxf(sqi + sq[j] - 2.f * G[(size_t)i * ld + j], 0.f);
      d = sqrtf(d2 + 1e-10f);
    }
    row[j] = d;
  }
  __syncthreads();

  int removed = 0;
  float s11 = 0.f, s30 = 0.f;
  while (removed < K_KNN) {
    float lmin = 3.4e38f;
    for (int j = tid; j < n; j += 256) lmin = fminf(lmin, row[j]);
    redv[tid] = lmin; __syncthreads();
    for (int s = 128; s > 0; s >>= 1) { if (tid < s) redv[tid] = fminf(redv[tid], redv[tid + s]); __syncthreads(); }
    const float vmin = redv[0];
    __syncthreads();
    int lc = 0;
    for (int j = tid; j < n; j += 256)
      if (row[j] == vmin) { row[j] = 3.4e38f; ++lc; }
    redc[tid] = lc; __syncthreads();
    for (int s = 128; s > 0; s >>= 1) { if (tid < s) redc[tid] += redc[tid + s]; __syncthreads(); }
    const int c = redc[0];
    __syncthreads();
    if (c == 0) break;
    if (removed <= KA        && KA        < removed + c) s11 = vmin;
    if (removed <= K_KNN - 1 && K_KNN - 1 < removed + c) s30 = vmin;
    removed += c;
  }

  const float inv  = 1.f / (s11 + 1e-10f);
  const float dn30 = s30 * inv;
  const float thr  = __expf(-dn30 * dn30);
  for (int j = tid; j < n; j += 256) {
    const float d2   = fmaxf(sqi + sq[j] - 2.f * G[(size_t)i * ld + j], 0.f);
    const float dist = sqrtf(d2 + 1e-10f) * inv;
    float a = __expf(-dist * dist);
    a = (a >= thr) ? a : 0.f;
    if (j == i) a = 1.f;
    G[(size_t)i * ld + j] = a;
  }
}

// rowsum[i] = sum_j aff[i][j] + aff[j][i]
__global__ __launch_bounds__(256)
void rowsum_kernel(const float* __restrict__ aff, float* __restrict__ rowsum, int n, int ld) {
  __shared__ float red[256];
  const int i = blockIdx.x, t = threadIdx.x;
  float s = 0.f;
  for (int j = t; j < n; j += 256)
    s += aff[(size_t)i * ld + j] + aff[(size_t)j * ld + i];
  red[t] = s; __syncthreads();
  for (int st = 128; st > 0; st >>= 1) { if (t < st) red[t] += red[t + st]; __syncthreads(); }
  if (t == 0) rowsum[i] = red[0];
}

// P[i][j] and P^T[j][i] = (aff[i][j]+aff[j][i]) / rowsum[i]
__global__ __launch_bounds__(256)
void normalize_kernel(const float* __restrict__ aff, const float* __restrict__ rowsum,
                      _Float16* __restrict__ P, _Float16* __restrict__ PT, int n, int ld) {
  const int j = blockIdx.x * 256 + threadIdx.x;
  const int i = blockIdx.y;
  if (j < n) {
    const float v = (aff[(size_t)i * ld + j] + aff[(size_t)j * ld + i]) / rowsum[i];
    P [(size_t)i * ld + j] = (_Float16)v;
    PT[(size_t)j * ld + i] = (_Float16)v;
  }
}

// hidden = mean over heads + LeakyReLU; emit f16 copy
__global__ void hidden_finalize(float* __restrict__ hidden, _Float16* __restrict__ h16, int total) {
  const int i = blockIdx.x * 256 + threadIdx.x;
  if (i < total) {
    float x = hidden[i] * (1.f / HEADS);
    x = (x >= 0.f) ? x : 0.01f * x;
    hidden[i] = x;
    h16[i]    = (_Float16)x;
  }
}

// cls = softmax(hidden @ Wc + bc), N=12: one block per row
__global__ __launch_bounds__(128)
void cls_kernel(const float* __restrict__ hidden, const float* __restrict__ Wc,
                const float* __restrict__ bc, float* __restrict__ out) {
  __shared__ float red[NCLUST][128];
  const int i = blockIdx.x, tid = threadIdx.x;
  float p[NCLUST];
#pragma unroll
  for (int c = 0; c < NCLUST; ++c) p[c] = 0.f;
  for (int k = tid; k < HID; k += 128) {
    const float h = hidden[(size_t)i * HID + k];
#pragma unroll
    for (int c = 0; c < NCLUST; ++c) p[c] += h * Wc[k * NCLUST + c];
  }
#pragma unroll
  for (int c = 0; c < NCLUST; ++c) red[c][tid] = p[c];
  __syncthreads();
  for (int s = 64; s > 0; s >>= 1) {
    if (tid < s)
#pragma unroll
      for (int c = 0; c < NCLUST; ++c) red[c][tid] += red[c][tid + s];
    __syncthreads();
  }
  if (tid == 0) {
    float lg[NCLUST], m = -3.4e38f;
#pragma unroll
    for (int c = 0; c < NCLUST; ++c) { lg[c] = red[c][0] + bc[c]; m = fmaxf(m, lg[c]); }
    float sum = 0.f;
#pragma unroll
    for (int c = 0; c < NCLUST; ++c) { lg[c] = __expf(lg[c] - m); sum += lg[c]; }
    const float inv = 1.f / sum;
#pragma unroll
    for (int c = 0; c < NCLUST; ++c) out[(size_t)i * NCLUST + c] = lg[c] * inv;
  }
}

// ---------------------------------------------------------------------------
// Host orchestration
// ---------------------------------------------------------------------------
extern "C" void kernel_launch(void* const* d_in, const int* in_sizes, int n_in,
                              void* d_out, int out_size, void* d_ws, size_t ws_size,
                              hipStream_t stream) {
  (void)in_sizes; (void)n_in; (void)out_size; (void)ws_size;
  const float* X   = (const float*)d_in[0];   // [NPTS, IN_DIM]
  const float* Wqk = (const float*)d_in[1];   // [H, IN_DIM, HID]
  const float* bqk = (const float*)d_in[2];   // [H, HID]
  const float* Wv  = (const float*)d_in[3];   // [H, IN_DIM, HID]
  const float* bv  = (const float*)d_in[4];   // [H, HID]
  const float* Wd  = (const float*)d_in[5];   // [HID, IN_DIM]
  const float* bd  = (const float*)d_in[6];   // [IN_DIM]
  const float* Wc  = (const float*)d_in[7];   // [HID, NCLUST]
  const float* bc  = (const float*)d_in[8];   // [NCLUST]
  // d_in[9] = t (static 6; matrix_power unrolled as P2, P4, P6=P4@P2)

  char* ws = (char*)d_ws;
  size_t off = 0;
  auto alloc = [&](size_t bytes) -> void* {
    off = (off + 255) & ~(size_t)255;
    void* p = ws + off;
    off += bytes;
    return p;
  };
  _Float16* X16   = (_Float16*)alloc((size_t)NP * KXP * sizeof(_Float16));
  _Float16* WqkT  = (_Float16*)alloc((size_t)HEADS * HID * KXP * sizeof(_Float16));
  _Float16* WvT   = (_Float16*)alloc((size_t)HEADS * HID * KXP * sizeof(_Float16));
  _Float16* WdT   = (_Float16*)alloc((size_t)KXP * HID * sizeof(_Float16));
  float*    bdp   = (float*)   alloc((size_t)KXP * sizeof(float));
  float*    G     = (float*)   alloc((size_t)NP * NP * sizeof(float));
  _Float16* Pa    = (_Float16*)alloc((size_t)NP * NP * sizeof(_Float16));
  _Float16* PaT   = (_Float16*)alloc((size_t)NP * NP * sizeof(_Float16));
  _Float16* Pb    = (_Float16*)alloc((size_t)NP * NP * sizeof(_Float16));
  _Float16* PbT   = (_Float16*)alloc((size_t)NP * NP * sizeof(_Float16));
  _Float16* Pc    = (_Float16*)alloc((size_t)NP * NP * sizeof(_Float16));
  _Float16* q16   = (_Float16*)alloc((size_t)NP * HID * sizeof(_Float16));
  _Float16* vT16  = (_Float16*)alloc((size_t)HID * NP * sizeof(_Float16));
  _Float16* h16   = (_Float16*)alloc((size_t)NP * HID * sizeof(_Float16));
  float*    hidden= (float*)   alloc((size_t)NP * HID * sizeof(float));
  float*    recon = (float*)   alloc((size_t)NP * KXP * sizeof(float));
  float*    sq    = (float*)   alloc((size_t)NPTS * sizeof(float));
  float*    rowsum= (float*)   alloc((size_t)NPTS * sizeof(float));

  auto cdiv = [](int a, int b) { return (a + b - 1) / b; };
  const dim3 blk(GT);
  const dim3 gNN(NP / BN, NP / BM);      // 24 x 24
  const dim3 gNH(HID / BN, NP / BM);     // 2 x 24
  const dim3 gND(KXP / BN, NP / BM);     // 16 x 24

  // ---- prep: padded f16 conversions (every call; deterministic) ----------
  pad16_kernel<<<dim3(KXP / 256, NP), 256, 0, stream>>>(X, X16, NPTS, IN_DIM, IN_DIM, KXP);
  for (int h = 0; h < HEADS; ++h) {
    padT16_kernel<<<dim3(KXP / 256, HID), 256, 0, stream>>>(
        Wqk + (size_t)h * IN_DIM * HID, WqkT + (size_t)h * HID * KXP, IN_DIM, HID, HID, KXP);
    padT16_kernel<<<dim3(KXP / 256, HID), 256, 0, stream>>>(
        Wv + (size_t)h * IN_DIM * HID, WvT + (size_t)h * HID * KXP, IN_DIM, HID, HID, KXP);
  }
  padT16_kernel<<<dim3(HID / 256, KXP), 256, 0, stream>>>(Wd, WdT, HID, IN_DIM, IN_DIM, HID);
  padbias_kernel<<<KXP / 256, 256, 0, stream>>>(bd, bdp, IN_DIM, KXP);
  zero16_kernel<<<4096, 256, 0, stream>>>(Pa,  (size_t)NP * NP);
  zero16_kernel<<<4096, 256, 0, stream>>>(PaT, (size_t)NP * NP);
  zero32_kernel<<<1024, 256, 0, stream>>>(hidden, (size_t)NP * HID);

  for (int h = 0; h < HEADS; ++h) {
    const _Float16* wqkT = WqkT + (size_t)h * HID * KXP;
    const _Float16* wvT  = WvT  + (size_t)h * HID * KXP;
    // q = X @ Wqk + bqk (f16);  vT = (X @ Wv + bv)^T (f16, vectorized CT store)
    wmma_gemm<EP_C16 | EP_BIAS><<<gNH, blk, 0, stream>>>(
        X16, wqkT, bqk + h * HID, q16, nullptr, nullptr, KXP, KXP, KXP, HID, 0);
    wmma_gemm<EP_CT16 | EP_BIAS><<<gNH, blk, 0, stream>>>(
        X16, wvT, bv + h * HID, nullptr, nullptr, vT16, KXP, KXP, KXP, 0, NP);
    // gram G = q @ q^T  (q16 serves as both A and Bt: [N][K] layout)
    sqnorm_kernel<<<NPTS, 256, 0, stream>>>(q16, sq, HID);
    wmma_gemm<EP_C32><<<gNN, blk, 0, stream>>>(
        q16, q16, nullptr, nullptr, G, nullptr, HID, HID, HID, NP, 0);
    // affinity (in place), symmetrize + row-normalize -> Pa, PaT
    markov_rows<<<NPTS, 256, 0, stream>>>(G, sq, NPTS, NP);
    rowsum_kernel<<<NPTS, 256, 0, stream>>>(G, rowsum, NPTS, NP);
    normalize_kernel<<<dim3(cdiv(NPTS, 256), NPTS), 256, 0, stream>>>(
        G, rowsum, Pa, PaT, NPTS, NP);
    // P^2 = P @ P     -> Pb, PbT (both layouts)
    wmma_gemm<EP_C16 | EP_CT16><<<gNN, blk, 0, stream>>>(
        Pa, PaT, nullptr, Pb, nullptr, PbT, NP, NP, NP, NP, NP);
    // P^4 = P2 @ P2   -> Pc
    wmma_gemm<EP_C16><<<gNN, blk, 0, stream>>>(
        Pb, PbT, nullptr, Pc, nullptr, nullptr, NP, NP, NP, NP, 0);
    // P^6 = P4 @ P2   -> Pa
    wmma_gemm<EP_C16><<<gNN, blk, 0, stream>>>(
        Pc, PbT, nullptr, Pa, nullptr, nullptr, NP, NP, NP, NP, 0);
    // hidden += P6 @ v
    wmma_gemm<EP_ACC32><<<gNH, blk, 0, stream>>>(
        Pa, vT16, nullptr, nullptr, hidden, nullptr, NP, NP, NP, HID, 0);
  }

  hidden_finalize<<<cdiv(NP * HID, 256), 256, 0, stream>>>(hidden, h16, NP * HID);

  // recon = softplus(hidden @ Wd + bd) -> padded ws, then exact copy to d_out
  wmma_gemm<EP_C32 | EP_BIAS | EP_SOFTPLUS><<<gND, blk, 0, stream>>>(
      h16, WdT, bdp, nullptr, recon, nullptr, HID, HID, HID, KXP, 0);
  copy_out_kernel<<<dim3(cdiv(IN_DIM, 256), NPTS), 256, 0, stream>>>(
      recon, (float*)d_out, IN_DIM, KXP);

  // cls = softmax(hidden @ Wc + bc) -> d_out tail
  cls_kernel<<<NPTS, 128, 0, stream>>>(hidden, Wc, bc,
                                       (float*)d_out + (size_t)NPTS * IN_DIM);
}
